// MultinomialDiffusion_20409684590741
// MI455X (gfx1250) — matrix operand order repaired and checked by reading, not compiled, not run on hardware
//
#include <hip/hip_runtime.h>
#include <hip/hip_bf16.h>
#include <math.h>
#include <stdint.h>

// ---------------- static problem dimensions (from reference) ----------------
#define BB     4096   // batch
#define DD     6425   // total one-hot width (25 * 257)
#define HH     1024   // hidden
#define TT     1000   // timesteps
#define NCOL   200    // ragged columns
#define DP     6432   // DD padded to multiple of 32 (K of GEMM1)
#define NP2    6464   // DD padded to multiple of 64 (N of GEMM2)

#define LDSTRIDE 40   // LDS tile row stride in elements (80B: bank-spread, 16B aligned)
#define ATILE (128 * LDSTRIDE)
#define BTILE (64 * LDSTRIDE)

typedef __attribute__((ext_vector_type(16))) __bf16 v16bf;
typedef __attribute__((ext_vector_type(8)))  float  v8f;

// ---------------- async global->LDS copy (CDNA5, ASYNCcnt-tracked) ----------------
// VDST VGPR holds the LDS byte address (low 32 bits of flat shared pointer per
// aperture rule); VADDR holds the 64-bit global address.
__device__ __forceinline__ void async_copy16(void* lds_ptr, const void* gptr) {
  unsigned loff = (unsigned)(uintptr_t)lds_ptr;
  asm volatile("global_load_async_to_lds_b128 %0, %1, off"
               :: "v"(loff), "v"(gptr) : "memory");
}
__device__ __forceinline__ void wait_async0() {
  asm volatile("s_wait_asynccnt 0x0" ::: "memory");
}

// ---------------- WMMA fragment loaders from LDS tiles (contiguous k order, same for A and B) ----------------
// A (16x32, MxK): lanes 0-15 -> M=lane, K=0..15; lanes 16-31 -> M=lane-16, K=16..31.
__device__ __forceinline__ v16bf lfrag_a(const __bf16* sA, int mloc, int lane) {
  const __bf16* p = sA + (mloc + (lane & 15)) * LDSTRIDE + ((lane >> 4) << 4);
  v16bf r;
#pragma unroll
  for (int j = 0; j < 16; ++j) r[j] = p[j];
  return r;
}
// B (32x16, KxN) from N-major tile: lanes 0-15 -> N=lane, K=0..15; lanes 16-31 -> K=16..31.
__device__ __forceinline__ v16bf lfrag_b(const __bf16* sB, int nloc, int lane) {
  const __bf16* p = sB + (nloc + (lane & 15)) * LDSTRIDE + ((lane >> 4) << 4);
  v16bf r;
#pragma unroll
  for (int j = 0; j < 16; ++j) r[j] = p[j];
  return r;
}

// ---------------- stage one K-step of workgroup tiles into LDS ----------------
// A tile: 128 rows x 32 elems (4 x 16B segs/row) = 512 xfers -> 2 per thread.
// B tile:  64 rows x 32 elems                    = 256 xfers -> 1 per thread.
__device__ __forceinline__ void stage_tiles(
    const __bf16* __restrict__ A, int lda, int m0b,
    const __bf16* __restrict__ BT, int ldb, int n0b,
    int kb, __bf16* da, __bf16* db, int tid) {
#pragma unroll
  for (int it = 0; it < 2; ++it) {
    int idx = tid + it * 256;
    int row = idx >> 2, seg = idx & 3;
    async_copy16(da + row * LDSTRIDE + seg * 8,
                 A + (size_t)(m0b + row) * lda + kb + seg * 8);
  }
  {
    int row = tid >> 2, seg = tid & 3;
    async_copy16(db + row * LDSTRIDE + seg * 8,
                 BT + (size_t)(n0b + row) * ldb + kb + seg * 8);
  }
}

// ---------------- double-buffered WMMA mainloop ----------------
__device__ __forceinline__ void gemm_core(
    const __bf16* __restrict__ A, int lda,
    const __bf16* __restrict__ BT, int ldb,
    int K, int m0b, int n0b,
    __bf16* sA, __bf16* sB,
    int tid, int lane, int wm, int wn, v8f (&acc)[2][2]) {
  stage_tiles(A, lda, m0b, BT, ldb, n0b, 0, sA, sB, tid);
  wait_async0();
  __syncthreads();
  int ib = 0;
  for (int kb = 0; kb < K; kb += 32, ib ^= 1) {
    __bf16* ca = sA + ib * ATILE;
    __bf16* cb = sB + ib * BTILE;
    if (kb + 32 < K) {  // overlap next tile's async DMA with this tile's WMMAs
      stage_tiles(A, lda, m0b, BT, ldb, n0b, kb + 32,
                  sA + (ib ^ 1) * ATILE, sB + (ib ^ 1) * BTILE, tid);
    }
    v16bf a0 = lfrag_a(ca, wm * 32,      lane);
    v16bf a1 = lfrag_a(ca, wm * 32 + 16, lane);
    v16bf g0 = lfrag_b(cb, wn * 32,      lane);
    v16bf g1 = lfrag_b(cb, wn * 32 + 16, lane);
    acc[0][0] = __builtin_amdgcn_wmma_f32_16x16x32_bf16(false, a0, false, g0, (short)0, acc[0][0], false, false);
    acc[0][1] = __builtin_amdgcn_wmma_f32_16x16x32_bf16(false, a0, false, g1, (short)0, acc[0][1], false, false);
    acc[1][0] = __builtin_amdgcn_wmma_f32_16x16x32_bf16(false, a1, false, g0, (short)0, acc[1][0], false, false);
    acc[1][1] = __builtin_amdgcn_wmma_f32_16x16x32_bf16(false, a1, false, g1, (short)0, acc[1][1], false, false);
    wait_async0();
    __syncthreads();
  }
}

// ---------------- conversion / packing kernels ----------------
__global__ void conv_A_kernel(const float* __restrict__ x, __bf16* __restrict__ y) {
  long i = (long)blockIdx.x * 256 + threadIdx.x;
  if (i >= (long)BB * DP) return;
  int row = (int)(i / DP), col = (int)(i % DP);
  float v = (col < DD) ? x[(size_t)row * DD + col] : 0.f;
  y[i] = (__bf16)v;
}
__global__ void conv_W1T_kernel(const float* __restrict__ w1, __bf16* __restrict__ y) {
  long i = (long)blockIdx.x * 256 + threadIdx.x;          // over [HH, DP]
  if (i >= (long)HH * DP) return;
  int h = (int)(i / DP), d = (int)(i % DP);
  float v = (d < DD) ? w1[(size_t)d * HH + h] : 0.f;      // W1 is [DD, HH]
  y[i] = (__bf16)v;
}
__global__ void conv_W2T_kernel(const float* __restrict__ w2, __bf16* __restrict__ y) {
  long i = (long)blockIdx.x * 256 + threadIdx.x;          // over [NP2, HH]
  if (i >= (long)NP2 * HH) return;
  int n = (int)(i / HH), k = (int)(i % HH);
  float v = (n < DD) ? w2[(size_t)k * DD + n] : 0.f;      // W2 is [HH, DD]
  y[i] = (__bf16)v;
}

// ---------------- diffusion schedule (fp64, matches reference init) ----------------
__global__ void sched_init_kernel(float* __restrict__ s) {
  if (threadIdx.x == 0 && blockIdx.x == 0) {
    float* la  = s;            // log_alpha[t]
    float* l1a = s + TT;       // log(1 - alpha[t])
    float* lc  = s + 2 * TT;   // cum log_alpha
    float* l1c = s + 3 * TT;   // log(1 - exp(cum))
    double cum = 0.0;
    for (int i = 0; i < TT; ++i) {
      double beta = 1e-4 + (0.02 - 1e-4) * (double)i / 999.0;
      double lga  = 0.5 * log(1.0 - beta);
      cum += lga;
      la[i]  = (float)lga;
      l1a[i] = (float)log(1.0 - exp(lga) + 1e-40);
      lc[i]  = (float)cum;
      l1c[i] = (float)log(1.0 - exp(cum) + 1e-40);
    }
  }
}

// ---------------- GEMM1: relu(log_x_t@W1 + temb[t] + Cont@Wc + b1) -> bf16 X ----------------
__global__ __launch_bounds__(256) void gemm1_kernel(
    const __bf16* __restrict__ A,    // [BB, DP] bf16 log_x_t
    const __bf16* __restrict__ BT,   // [HH, DP] bf16 W1^T
    const float*  __restrict__ Cont, // [BB, 16]
    const float*  __restrict__ Wc,   // [16, HH]
    const float*  __restrict__ temb, // [TT, HH]
    const float*  __restrict__ bias1,// [HH]
    const int*    __restrict__ ts,   // [BB]
    __bf16*       __restrict__ X)    // [BB, HH]
{
  __shared__ __bf16 sA[2 * ATILE];
  __shared__ __bf16 sB[2 * BTILE];
  const int tid = threadIdx.x;
  const int lane = tid & 31, wave = tid >> 5;
  const int wm = wave >> 1, wn = wave & 1;
  const int m0b = blockIdx.x * 128, n0b = blockIdx.y * 64;
  v8f acc[2][2] = {};
  gemm_core(A, DP, BT, DP, DP, m0b, n0b, sA, sB, tid, lane, wm, wn, acc);

  const int m0 = m0b + wm * 32, n0 = n0b + wn * 32;
  const int nlo = lane & 15, mhi = (lane >> 4) << 3;
#pragma unroll
  for (int mi = 0; mi < 2; ++mi)
#pragma unroll
  for (int ni = 0; ni < 2; ++ni) {
#pragma unroll
    for (int r = 0; r < 8; ++r) {
      const int M = m0 + 16 * mi + mhi + r;
      const int N = n0 + 16 * ni + nlo;
      float v = acc[mi][ni][r];
      v += temb[(size_t)ts[M] * HH + N] + bias1[N];
      float dc = 0.f;
#pragma unroll
      for (int k = 0; k < 16; ++k) dc += Cont[M * 16 + k] * Wc[k * HH + N];
      v += dc;
      v = v > 0.f ? v : 0.f;
      X[(size_t)M * HH + N] = (__bf16)v;
    }
  }
}

// ---------------- GEMM2: X@W2 + b2 -> fp32 Out [BB, NP2] ----------------
__global__ __launch_bounds__(256) void gemm2_kernel(
    const __bf16* __restrict__ A,    // [BB, HH] bf16 hidden
    const __bf16* __restrict__ BT,   // [NP2, HH] bf16 W2^T (zero-padded rows)
    const float*  __restrict__ b2,   // [DD]
    float*        __restrict__ Out)  // [BB, NP2]
{
  __shared__ __bf16 sA[2 * ATILE];
  __shared__ __bf16 sB[2 * BTILE];
  const int tid = threadIdx.x;
  const int lane = tid & 31, wave = tid >> 5;
  const int wm = wave >> 1, wn = wave & 1;
  const int m0b = blockIdx.x * 128, n0b = blockIdx.y * 64;
  v8f acc[2][2] = {};
  gemm_core(A, HH, BT, HH, HH, m0b, n0b, sA, sB, tid, lane, wm, wn, acc);

  const int m0 = m0b + wm * 32, n0 = n0b + wn * 32;
  const int nlo = lane & 15, mhi = (lane >> 4) << 3;
#pragma unroll
  for (int mi = 0; mi < 2; ++mi)
#pragma unroll
  for (int ni = 0; ni < 2; ++ni) {
#pragma unroll
    for (int r = 0; r < 8; ++r) {
      const int M = m0 + 16 * mi + mhi + r;
      const int N = n0 + 16 * ni + nlo;
      float v = acc[mi][ni][r] + (N < DD ? b2[N] : 0.f);
      Out[(size_t)M * NP2 + N] = v;
    }
  }
}

// ---------------- per-sample loss: q_posterior x2, ragged LSE, KL / NLL ----------------
__device__ __forceinline__ float lae(float a, float b) {   // logaddexp
  float m = fmaxf(a, b);
  return m + log1pf(__expf(fminf(a, b) - m));
}

__global__ __launch_bounds__(256) void loss_kernel(
    const float* __restrict__ xs_all,  // log_x_start [BB, DD]
    const float* __restrict__ xt_all,  // log_x_t     [BB, DD]
    const float* __restrict__ Out,     // MLP output  [BB, NP2]
    const int*   __restrict__ ts,      // [BB]
    const float* __restrict__ sched,   // 4*TT
    float*       __restrict__ loss)    // [BB]
{
  __shared__ float sKL[256], sDE[256];
  const int b = blockIdx.x;
  const int t = ts[b];
  const int tm1 = t > 0 ? t - 1 : 0;
  const float la  = sched[t];
  const float l1a = sched[TT + t];
  const float lc  = sched[2 * TT + tm1];
  const float l1c = sched[3 * TT + tm1];
  const bool  is0 = (t == 0);

  const int c = threadIdx.x;
  float klc = 0.f, dec = 0.f;
  if (c < NCOL) {
    const int szs[8] = {2, 3, 4, 8, 16, 32, 64, 128};
    const int prf[8] = {0, 2, 5, 9, 17, 33, 65, 129};
    const int cnt = szs[c & 7];
    const int off = (c >> 3) * 257 + prf[c & 7];
    const float lk = __logf((float)cnt);
    const float* o  = Out    + (size_t)b * NP2 + off;
    const float* xt = xt_all + (size_t)b * DD  + off;
    const float* xs = xs_all + (size_t)b * DD  + off;

    // per-column log-softmax denominator of MLP output
    float m1 = -1e30f;
    for (int j = 0; j < cnt; ++j) m1 = fmaxf(m1, o[j]);
    float s1 = 0.f;
    for (int j = 0; j < cnt; ++j) s1 += __expf(o[j] - m1);
    const float lseO = m1 + __logf(s1);

    // pass 1: maxima of both unnormalized posteriors
    float mT = -1e30f, mM = -1e30f;
    for (int j = 0; j < cnt; ++j) {
      float q1 = lae(xt[j] + la, l1a - lk);
      float ut = (is0 ? xs[j] : lae(xs[j] + lc, l1c - lk)) + q1;
      float rc = o[j] - lseO;
      float um = (is0 ? rc : lae(rc + lc, l1c - lk)) + q1;
      mT = fmaxf(mT, ut); mM = fmaxf(mM, um);
    }
    // pass 2: segment sums
    float sT = 0.f, sM = 0.f;
    for (int j = 0; j < cnt; ++j) {
      float q1 = lae(xt[j] + la, l1a - lk);
      float ut = (is0 ? xs[j] : lae(xs[j] + lc, l1c - lk)) + q1;
      float rc = o[j] - lseO;
      float um = (is0 ? rc : lae(rc + lc, l1c - lk)) + q1;
      sT += __expf(ut - mT); sM += __expf(um - mM);
    }
    const float lseT = mT + __logf(sT), lseM = mM + __logf(sM);
    // pass 3: KL and decoder-NLL accumulation
    for (int j = 0; j < cnt; ++j) {
      float q1 = lae(xt[j] + la, l1a - lk);
      float ut = (is0 ? xs[j] : lae(xs[j] + lc, l1c - lk)) + q1;
      float rc = o[j] - lseO;
      float um = (is0 ? rc : lae(rc + lc, l1c - lk)) + q1;
      float lt = ut - lseT, lm = um - lseM;
      klc += __expf(lt) * (lt - lm);
      dec += __expf(xs[j]) * lm;
    }
    klc /= (float)cnt;
    dec = -dec / (float)cnt;
  }
  sKL[c] = klc; sDE[c] = dec;
  __syncthreads();
  for (int s = 128; s > 0; s >>= 1) {
    if (c < s) { sKL[c] += sKL[c + s]; sDE[c] += sDE[c + s]; }
    __syncthreads();
  }
  if (c == 0) {
    float klm = sKL[0] / (float)NCOL;
    float dem = sDE[0] / (float)NCOL;
    loss[b] = is0 ? dem : klm;
  }
}

// ---------------- launch ----------------
extern "C" void kernel_launch(void* const* d_in, const int* in_sizes, int n_in,
                              void* d_out, int out_size, void* d_ws, size_t ws_size,
                              hipStream_t stream) {
  (void)in_sizes; (void)n_in; (void)out_size; (void)ws_size;
  const float* log_x_start = (const float*)d_in[0];
  const float* log_x_t     = (const float*)d_in[1];
  const float* Cont        = (const float*)d_in[2];
  const float* W1          = (const float*)d_in[3];
  const float* Wc          = (const float*)d_in[4];
  const float* temb        = (const float*)d_in[5];
  const float* b1          = (const float*)d_in[6];
  const float* W2          = (const float*)d_in[7];
  const float* b2          = (const float*)d_in[8];
  const int*   ts          = (const int*)d_in[9];
  float* loss = (float*)d_out;

  char* ws = (char*)d_ws;
  size_t off = 0;
  auto take = [&](size_t bytes) -> char* {
    char* p = ws + off;
    off += (bytes + 255) & ~(size_t)255;
    return p;
  };
  __bf16* Abf   = (__bf16*)take((size_t)BB  * DP  * 2);   // bf16 log_x_t, K-padded
  __bf16* W1T   = (__bf16*)take((size_t)HH  * DP  * 2);   // bf16 W1^T
  __bf16* W2T   = (__bf16*)take((size_t)NP2 * HH  * 2);   // bf16 W2^T, N-padded
  __bf16* Xbf   = (__bf16*)take((size_t)BB  * HH  * 2);   // bf16 hidden
  float*  Out   = (float*) take((size_t)BB  * NP2 * 4);   // fp32 MLP output
  float*  sched = (float*) take((size_t)4 * TT * 4);      // diffusion schedule

  sched_init_kernel<<<1, 1, 0, stream>>>(sched);
  {
    long n = (long)BB * DP;
    conv_A_kernel<<<(unsigned)((n + 255) / 256), 256, 0, stream>>>(log_x_t, Abf);
  }
  {
    long n = (long)HH * DP;
    conv_W1T_kernel<<<(unsigned)((n + 255) / 256), 256, 0, stream>>>(W1, W1T);
  }
  {
    long n = (long)NP2 * HH;
    conv_W2T_kernel<<<(unsigned)((n + 255) / 256), 256, 0, stream>>>(W2, W2T);
  }
  gemm1_kernel<<<dim3(BB / 128, HH / 64), 256, 0, stream>>>(Abf, W1T, Cont, Wc, temb, b1, ts, Xbf);
  gemm2_kernel<<<dim3(BB / 128, NP2 / 64), 256, 0, stream>>>(Xbf, W2T, b2, Out);
  loss_kernel<<<BB, 256, 0, stream>>>(log_x_start, log_x_t, Out, ts, sched, loss);
}
